// AttentionLayer_17308718203232
// MI455X (gfx1250) — matrix operand order repaired and checked by reading, not compiled, not run on hardware
//
#include <hip/hip_runtime.h>
#include <hip/hip_bf16.h>

typedef __attribute__((ext_vector_type(8)))  __bf16 v8bf;
typedef __attribute__((ext_vector_type(16))) __bf16 v16bf;
typedef __attribute__((ext_vector_type(8)))  float  v8f;

#define BB 8
#define SS 2048
#define DD 512
#define MM (BB * SS)   // 16384 rows

// ---------------- conversion kernels ----------------

// fp32 -> bf16, 8 elements per thread
__global__ void cvt_x_kernel(const float* __restrict__ x, __bf16* __restrict__ xb) {
    int i = (blockIdx.x * blockDim.x + threadIdx.x) * 8;
#pragma unroll
    for (int j = 0; j < 8; ++j) xb[i + j] = (__bf16)x[i + j];
}

// W[k][n] (fp32, DxD) -> Wt[n][k] (bf16), for both Wk and Wv
__global__ void cvt_w_kernel(const float* __restrict__ Wk, const float* __restrict__ Wv,
                             __bf16* __restrict__ wkT, __bf16* __restrict__ wvT) {
    int idx = blockIdx.x * blockDim.x + threadIdx.x;   // 0 .. D*D-1
    int n = idx / DD;
    int k = idx % DD;
    wkT[n * DD + k] = (__bf16)Wk[k * DD + n];
    wvT[n * DD + k] = (__bf16)Wv[k * DD + n];
}

// ---------------- fused K/V projection GEMM (WMMA bf16, LDS-staged B) ----------------
// C[M,N] = A[M,K] * W[K,N] + bias, for Wk and Wv simultaneously (A reuse).
// Block: 256 threads = 8 waves; wave computes 16 rows x 64 cols.
// B panels (64 cols x 512 K, both matrices) staged in LDS via async-to-LDS DMA.
#define LDS_STRIDE 520   // bf16 elems per LDS row: 512 + 8 pad (260 dwords, bank-friendly)

__global__ __launch_bounds__(256)
void gemm_kv_kernel(const __bf16* __restrict__ xb,
                    const __bf16* __restrict__ wkT, const __bf16* __restrict__ wvT,
                    const float* __restrict__ bk, const float* __restrict__ bv,
                    float* __restrict__ Kmat, float* __restrict__ Vmat) {
    __shared__ __align__(16) __bf16 ldsK[64 * LDS_STRIDE];
    __shared__ __align__(16) __bf16 ldsV[64 * LDS_STRIDE];

    const int tid  = threadIdx.x;
    const int lane = tid & 31;
    const int wave = tid >> 5;
    const int l  = lane & 15;
    const int hi = lane >> 4;

    const int m  = blockIdx.x * 128 + wave * 16;   // row tile base
    const int n0 = blockIdx.y * 64;                // col tile base

    // ---- async-stage B panels: 64 rows x 1024B each, 4096 16B-chunks/panel ----
    {
        const unsigned ldsKbase = (unsigned)(unsigned long long)&ldsK[0];
        const unsigned ldsVbase = (unsigned)(unsigned long long)&ldsV[0];
        const void* gk = (const void*)(wkT + (size_t)n0 * DD);
        const void* gv = (const void*)(wvT + (size_t)n0 * DD);
#pragma unroll
        for (int i = 0; i < 16; ++i) {
            int c    = tid + i * 256;        // chunk id 0..4095
            int row  = c >> 6;               // 0..63
            int cib  = (c & 63) * 16;        // byte offset within row
            unsigned voff = (unsigned)(row * DD * 2 + cib);          // global byte off
            unsigned loff = (unsigned)(row * LDS_STRIDE * 2 + cib);  // lds byte off
            unsigned ldsKaddr = ldsKbase + loff;
            unsigned ldsVaddr = ldsVbase + loff;
            asm volatile("global_load_async_to_lds_b128 %0, %1, %2"
                         :: "v"(ldsKaddr), "v"(voff), "s"(gk) : "memory");
            asm volatile("global_load_async_to_lds_b128 %0, %1, %2"
                         :: "v"(ldsVaddr), "v"(voff), "s"(gv) : "memory");
        }
        asm volatile("s_wait_asynccnt 0x0" ::: "memory");
    }
    __syncthreads();

    v8f accK[4];
    v8f accV[4];
#pragma unroll
    for (int nt = 0; nt < 4; ++nt) {
        accK[nt] = (v8f)(0.0f);
        accV[nt] = (v8f)(0.0f);
    }

    const __bf16* arow = xb + (size_t)(m + l) * DD;

    // software-pipelined A fragment (one k-step ahead)
    v8bf alo = *(const v8bf*)(arow + 16 * hi);
    v8bf ahi = *(const v8bf*)(arow + 16 * hi + 8);

    for (int kk = 0; kk < DD; kk += 32) {
        const int kn = (kk + 32 < DD) ? (kk + 32) : 0;   // wrap-guarded prefetch
        v8bf nlo = *(const v8bf*)(arow + kn + 16 * hi);
        v8bf nhi = *(const v8bf*)(arow + kn + 16 * hi + 8);

        v16bf a;
#pragma unroll
        for (int i = 0; i < 8; ++i) { a[i] = alo[i]; a[8 + i] = ahi[i]; }

#pragma unroll
        for (int nt = 0; nt < 4; ++nt) {
            const int ln = nt * 16 + l;                  // 0..63 panel row
            {
                const __bf16* brow = &ldsK[ln * LDS_STRIDE + kk + 16 * hi];
                v8bf blo = *(const v8bf*)(brow);
                v8bf bhi = *(const v8bf*)(brow + 8);
                v16bf bfr;
#pragma unroll
                for (int i = 0; i < 8; ++i) { bfr[i] = blo[i]; bfr[8 + i] = bhi[i]; }
                accK[nt] = __builtin_amdgcn_wmma_f32_16x16x32_bf16(
                    false, a, false, bfr, (short)0, accK[nt], false, false);
            }
            {
                const __bf16* brow = &ldsV[ln * LDS_STRIDE + kk + 16 * hi];
                v8bf blo = *(const v8bf*)(brow);
                v8bf bhi = *(const v8bf*)(brow + 8);
                v16bf bfr;
#pragma unroll
                for (int i = 0; i < 8; ++i) { bfr[i] = blo[i]; bfr[8 + i] = bhi[i]; }
                accV[nt] = __builtin_amdgcn_wmma_f32_16x16x32_bf16(
                    false, a, false, bfr, (short)0, accV[nt], false, false);
            }
        }
        alo = nlo;
        ahi = nhi;
    }

    // Epilogue: D-matrix layout -> row = m + r + 8*hi, col = n0 + 16*nt + l
#pragma unroll
    for (int nt = 0; nt < 4; ++nt) {
        const int n = n0 + nt * 16 + l;
        const float biasK = bk[n];
        const float biasV = bv[n];
#pragma unroll
        for (int r = 0; r < 8; ++r) {
            const int row = m + r + 8 * hi;
            Kmat[(size_t)row * DD + n] = accK[nt][r] + biasK;
            Vmat[(size_t)row * DD + n] = accV[nt][r] + biasV;
        }
    }
}

// ---------------- q_last projection (tiny GEMV, fp32) ----------------
__global__ void qproj_kernel(const float* __restrict__ x, const float* __restrict__ Wq,
                             const float* __restrict__ bq, float* __restrict__ qlast) {
    const int b = blockIdx.x;
    const int e = threadIdx.x;
    const float* xr = x + ((size_t)b * SS + (SS - 1)) * DD;
    float acc = bq[e];
    for (int d = 0; d < DD; ++d) acc += xr[d] * Wq[(size_t)d * DD + e];
    qlast[b * DD + e] = acc;
}

// ---------------- scores + softmax weights ----------------
__global__ void softmax_kernel(const float* __restrict__ qlast, const float* __restrict__ Kmat,
                               float* __restrict__ wts) {
    const int b = blockIdx.x;
    const int tid = threadIdx.x;
    __shared__ float qs[DD];
    __shared__ float red[256];

    qs[tid] = qlast[b * DD + tid];
    qs[tid + 256] = qlast[b * DD + tid + 256];
    __syncthreads();

    const float scale = 0.044194173824159216f;  // 1/sqrt(512)
    float sc[8];
    float lmax = -3.0e38f;
#pragma unroll
    for (int j = 0; j < 8; ++j) {
        const int s = tid + j * 256;
        const float* krow = Kmat + ((size_t)b * SS + s) * DD;
        float acc = 0.0f;
        for (int d = 0; d < DD; ++d) acc += qs[d] * krow[d];
        sc[j] = acc * scale;
        lmax = fmaxf(lmax, sc[j]);
    }

    red[tid] = lmax;
    __syncthreads();
    for (int off = 128; off > 0; off >>= 1) {
        if (tid < off) red[tid] = fmaxf(red[tid], red[tid + off]);
        __syncthreads();
    }
    const float gmax = red[0];
    __syncthreads();

    float lsum = 0.0f;
#pragma unroll
    for (int j = 0; j < 8; ++j) {
        sc[j] = __expf(sc[j] - gmax);
        lsum += sc[j];
    }
    red[tid] = lsum;
    __syncthreads();
    for (int off = 128; off > 0; off >>= 1) {
        if (tid < off) red[tid] += red[tid + off];
        __syncthreads();
    }
    const float inv = 1.0f / red[0];

#pragma unroll
    for (int j = 0; j < 8; ++j) wts[b * SS + tid + j * 256] = sc[j] * inv;
}

// ---------------- weighted sum over V ----------------
__global__ void attn_out_kernel(const float* __restrict__ wts, const float* __restrict__ Vmat,
                                float* __restrict__ out) {
    const int b = blockIdx.x;
    const int d = threadIdx.x;
    float acc = 0.0f;
    const float* vbase = Vmat + (size_t)b * SS * DD + d;
    const float* wbase = wts + b * SS;
    for (int s = 0; s < SS; ++s) acc += wbase[s] * vbase[(size_t)s * DD];
    out[b * DD + d] = acc;
}

// ---------------- launcher ----------------
extern "C" void kernel_launch(void* const* d_in, const int* in_sizes, int n_in,
                              void* d_out, int out_size, void* d_ws, size_t ws_size,
                              hipStream_t stream) {
    const float* x  = (const float*)d_in[0];
    const float* Wq = (const float*)d_in[1];
    const float* bq = (const float*)d_in[2];
    const float* Wk = (const float*)d_in[3];
    const float* bk = (const float*)d_in[4];
    const float* Wv = (const float*)d_in[5];
    const float* bv = (const float*)d_in[6];
    float* out = (float*)d_out;

    char* ws = (char*)d_ws;
    size_t off = 0;
    __bf16* xb  = (__bf16*)(ws + off); off += (size_t)MM * DD * sizeof(__bf16);   // 16 MB
    __bf16* wkT = (__bf16*)(ws + off); off += (size_t)DD * DD * sizeof(__bf16);   // 0.5 MB
    __bf16* wvT = (__bf16*)(ws + off); off += (size_t)DD * DD * sizeof(__bf16);   // 0.5 MB
    float* Kmat = (float*)(ws + off);  off += (size_t)MM * DD * sizeof(float);    // 32 MB
    float* Vmat = (float*)(ws + off);  off += (size_t)MM * DD * sizeof(float);    // 32 MB
    float* qlast = (float*)(ws + off); off += (size_t)BB * DD * sizeof(float);
    float* wts   = (float*)(ws + off); off += (size_t)BB * SS * sizeof(float);

    cvt_x_kernel<<<(MM * DD / 8 + 255) / 256, 256, 0, stream>>>(x, xb);
    cvt_w_kernel<<<(DD * DD + 255) / 256, 256, 0, stream>>>(Wk, Wv, wkT, wvT);
    gemm_kv_kernel<<<dim3(MM / 128, DD / 64), 256, 0, stream>>>(xb, wkT, wvT, bk, bv, Kmat, Vmat);
    qproj_kernel<<<BB, DD, 0, stream>>>(x, Wq, bq, qlast);
    softmax_kernel<<<BB, 256, 0, stream>>>(qlast, Kmat, wts);
    attn_out_kernel<<<BB, DD, 0, stream>>>(wts, Vmat, out);
}
